// PromptCrossAttention_48112223650142
// MI455X (gfx1250) — matrix operand order repaired and checked by reading, not compiled, not run on hardware
//
#include <hip/hip_runtime.h>

// ---------------------------------------------------------------------------
// PromptCrossAttention on MI455X (gfx1250, wave32, WMMA).
// Both 512x512x4096 batched GEMMs run on v_wmma_f32_16x16x32_bf16
// (bf16 operands / f32 accumulation) with 32x64 register tiles per wave
// (1.5 b128 loads per WMMA); conv, softmax, top-k, residual in f32.
// ---------------------------------------------------------------------------

#define BB  8
#define CC  512
#define HH  64
#define WW  64
#define HWN (HH * WW)      // 4096
#define KEEP 51            // max(1, int(0.1*512))

typedef __attribute__((ext_vector_type(16))) __bf16 bf16x16;
typedef __attribute__((ext_vector_type(8)))  __bf16 bf16x8;
typedef __attribute__((ext_vector_type(8)))  float  f32x8;

__device__ __forceinline__ __bf16 f2bf(float f) {
    unsigned u = __builtin_bit_cast(unsigned, f);
    unsigned r = u + 0x7FFFu + ((u >> 16) & 1u);   // round-to-nearest-even
    unsigned short h = (unsigned short)(r >> 16);
    return __builtin_bit_cast(__bf16, h);
}

// ---------------------------------------------------------------------------
// Depthwise 3x3 conv (SAME padding) + bias; emit bf16, optionally transposed
// to (B, HW, C) so the attn*V GEMM's B-fragments load contiguously.
// ---------------------------------------------------------------------------
__global__ void dwconv3x3_bf16(const float* __restrict__ x,
                               const float* __restrict__ w9,
                               const float* __restrict__ bias,
                               __bf16* __restrict__ y, int transposed) {
    int idx = blockIdx.x * blockDim.x + threadIdx.x;
    if (idx >= BB * CC * HWN) return;
    int hw = idx % HWN;
    int c  = (idx / HWN) % CC;
    int b  = idx / (HWN * CC);
    int h  = hw / WW, ww = hw % WW;

    const float* xp = x + ((size_t)b * CC + c) * HWN;
    const float* wp = w9 + c * 9;
    float acc = bias[c];
#pragma unroll
    for (int dy = 0; dy < 3; ++dy) {
        int yy = h + dy - 1;
        if (yy < 0 || yy >= HH) continue;
#pragma unroll
        for (int dx = 0; dx < 3; ++dx) {
            int xx = ww + dx - 1;
            if (xx < 0 || xx >= WW) continue;
            acc += xp[yy * WW + xx] * wp[dy * 3 + dx];
        }
    }
    size_t o = transposed ? ((size_t)b * HWN + hw) * CC + c
                          : ((size_t)b * CC + c) * HWN + hw;
    y[o] = f2bf(acc);
}

// ---------------------------------------------------------------------------
// GEMM1: attn[b][m][n] = (1/sqrt(C)) * sum_K q[b][m][K] * k[b][n][K]
// One wave per 32x64 output tile: 2 A frags x 4 B frags -> 8 WMMAs / k-step.
// grid = (CC/64, CC/32, BB), block = 32.
// ---------------------------------------------------------------------------
__global__ void gemm_qk_wmma(const __bf16* __restrict__ q,
                             const __bf16* __restrict__ k,
                             float* __restrict__ attn) {
    const int lane = threadIdx.x;
    const int b  = blockIdx.z;
    const int m0 = blockIdx.y * 32;
    const int n0 = blockIdx.x * 64;
    const int mr = lane & 15;
    const int kh = lane >> 4;

    const __bf16* ar[2];
    ar[0] = q + (size_t)(b * CC + m0 +  0 + mr) * HWN;
    ar[1] = q + (size_t)(b * CC + m0 + 16 + mr) * HWN;
    const __bf16* br[4];
#pragma unroll
    for (int t = 0; t < 4; ++t)
        br[t] = k + (size_t)(b * CC + n0 + t * 16 + mr) * HWN;

    f32x8 acc[2][4];
#pragma unroll
    for (int a = 0; a < 2; ++a)
#pragma unroll
        for (int t = 0; t < 4; ++t) acc[a][t] = (f32x8){};

    for (int kk = 0; kk < HWN; kk += 32) {
        bf16x16 A[2];
#pragma unroll
        for (int a = 0; a < 2; ++a) {
            bf16x8 alo = *(const bf16x8*)(ar[a] + kk + kh * 16);
            bf16x8 ahi = *(const bf16x8*)(ar[a] + kk + kh * 16 + 8);
            A[a] = __builtin_shufflevector(alo, ahi,
                0,1,2,3,4,5,6,7,8,9,10,11,12,13,14,15);
        }
        bf16x16 Bf[4];
#pragma unroll
        for (int t = 0; t < 4; ++t)
            Bf[t] = *(const bf16x16*)(br[t] + kk + kh * 16);
#pragma unroll
        for (int a = 0; a < 2; ++a)
#pragma unroll
            for (int t = 0; t < 4; ++t)
                acc[a][t] = __builtin_amdgcn_wmma_f32_16x16x32_bf16(
                    false, A[a], false, Bf[t], (short)0, acc[a][t], false, false);
    }
    const float scale = 0.04419417382415922f; // 1/sqrt(512)
    float* op = attn + (size_t)b * CC * CC;
#pragma unroll
    for (int a = 0; a < 2; ++a)
#pragma unroll
        for (int t = 0; t < 4; ++t)
#pragma unroll
            for (int r = 0; r < 8; ++r) {
                int m = m0 + a * 16 + kh * 8 + r;
                op[(size_t)m * CC + n0 + t * 16 + mr] = acc[a][t][r] * scale;
            }
}

// ---------------------------------------------------------------------------
// Softmax over last dim (rows of length 512). One wave per row.
// ---------------------------------------------------------------------------
__global__ void softmax_rows(float* __restrict__ attn) {
    int lane = threadIdx.x;
    float* p = attn + (size_t)blockIdx.x * CC;
    float vals[16];
    float mx = -3.402823466e38f;
#pragma unroll
    for (int e = 0; e < 16; ++e) { vals[e] = p[lane + e * 32]; mx = fmaxf(mx, vals[e]); }
#pragma unroll
    for (int off = 16; off >= 1; off >>= 1) mx = fmaxf(mx, __shfl_xor(mx, off, 32));
    float sum = 0.f;
#pragma unroll
    for (int e = 0; e < 16; ++e) { vals[e] = __expf(vals[e] - mx); sum += vals[e]; }
#pragma unroll
    for (int off = 16; off >= 1; off >>= 1) sum += __shfl_xor(sum, off, 32);
    float inv = 1.0f / sum;
#pragma unroll
    for (int e = 0; e < 16; ++e) p[lane + e * 32] = vals[e] * inv;
}

// ---------------------------------------------------------------------------
// Top-KEEP along rows (dim=1), per column. One wave per (b, col).
// Bitwise binary search on sortable keys for the KEEP-th largest; ties kept
// in ascending row order (matches stable argsort of -attn).
// Writes masked f32 to d_out and bf16 copy to ws for GEMM2.
// ---------------------------------------------------------------------------
__global__ void topk_mask(const float* __restrict__ attn,
                          float* __restrict__ attn_out,
                          __bf16* __restrict__ attnm) {
    int lane = threadIdx.x;
    int col  = blockIdx.x;
    int b    = blockIdx.y;
    const float* base = attn + (size_t)b * CC * CC + col;

    float vals[16]; unsigned keys[16];
#pragma unroll
    for (int e = 0; e < 16; ++e) {
        int r = e * 32 + lane;
        float v = base[(size_t)r * CC];
        vals[e] = v;
        unsigned u = __builtin_bit_cast(unsigned, v);
        keys[e] = (u & 0x80000000u) ? ~u : (u | 0x80000000u);
    }
    unsigned thr = 0;
    for (int bit = 31; bit >= 0; --bit) {
        unsigned cand = thr | (1u << bit);
        int cnt = 0;
#pragma unroll
        for (int e = 0; e < 16; ++e) cnt += (keys[e] >= cand) ? 1 : 0;
#pragma unroll
        for (int off = 16; off >= 1; off >>= 1) cnt += __shfl_xor(cnt, off, 32);
        if (cnt >= KEEP) thr = cand;
    }
    int ngt = 0;
#pragma unroll
    for (int e = 0; e < 16; ++e) ngt += (keys[e] > thr) ? 1 : 0;
#pragma unroll
    for (int off = 16; off >= 1; off >>= 1) ngt += __shfl_xor(ngt, off, 32);
    int quota = KEEP - ngt;

    unsigned lmask = lane ? (0xFFFFFFFFu >> (32 - lane)) : 0u;
    int running = 0;
    float*  outp = attn_out + (size_t)b * CC * CC + col;
    __bf16* mp   = attnm   + (size_t)b * CC * CC + col;
#pragma unroll
    for (int e = 0; e < 16; ++e) {
        bool eq = (keys[e] == thr);
        unsigned bl = __builtin_amdgcn_ballot_w32(eq);
        int rank = __builtin_popcount(bl & lmask);
        bool keepme = (keys[e] > thr) || (eq && (running + rank) < quota);
        running += __builtin_popcount(bl);
        float ov = keepme ? vals[e] : 0.0f;
        int r = e * 32 + lane;
        outp[(size_t)r * CC] = ov;
        mp[(size_t)r * CC]   = f2bf(ov);
    }
}

// ---------------------------------------------------------------------------
// GEMM2: out[b][m][n] = x_q[b][m][n] + sum_d attnm[b][m][d] * vT[b][n][d]
// vT is (B, HW, C) so B-fragments are contiguous. One wave per 32x64 tile.
// grid = (HWN/64, CC/32, BB), block = 32.
// ---------------------------------------------------------------------------
__global__ void gemm_av_wmma(const __bf16* __restrict__ attnm,
                             const __bf16* __restrict__ vT,
                             const float* __restrict__ xq,
                             float* __restrict__ outp) {
    const int lane = threadIdx.x;
    const int b  = blockIdx.z;
    const int m0 = blockIdx.y * 32;
    const int n0 = blockIdx.x * 64;
    const int mr = lane & 15;
    const int kh = lane >> 4;

    const __bf16* ar[2];
    ar[0] = attnm + (size_t)b * CC * CC + (size_t)(m0 +  0 + mr) * CC;
    ar[1] = attnm + (size_t)b * CC * CC + (size_t)(m0 + 16 + mr) * CC;
    const __bf16* br[4];
#pragma unroll
    for (int t = 0; t < 4; ++t)
        br[t] = vT + ((size_t)b * HWN + n0 + t * 16 + mr) * CC;

    f32x8 acc[2][4];
#pragma unroll
    for (int a = 0; a < 2; ++a)
#pragma unroll
        for (int t = 0; t < 4; ++t) acc[a][t] = (f32x8){};

    for (int kk = 0; kk < CC; kk += 32) {
        bf16x16 A[2];
#pragma unroll
        for (int a = 0; a < 2; ++a) {
            bf16x8 alo = *(const bf16x8*)(ar[a] + kk + kh * 16);
            bf16x8 ahi = *(const bf16x8*)(ar[a] + kk + kh * 16 + 8);
            A[a] = __builtin_shufflevector(alo, ahi,
                0,1,2,3,4,5,6,7,8,9,10,11,12,13,14,15);
        }
        bf16x16 Bf[4];
#pragma unroll
        for (int t = 0; t < 4; ++t)
            Bf[t] = *(const bf16x16*)(br[t] + kk + kh * 16);
#pragma unroll
        for (int a = 0; a < 2; ++a)
#pragma unroll
            for (int t = 0; t < 4; ++t)
                acc[a][t] = __builtin_amdgcn_wmma_f32_16x16x32_bf16(
                    false, A[a], false, Bf[t], (short)0, acc[a][t], false, false);
    }
#pragma unroll
    for (int a = 0; a < 2; ++a)
#pragma unroll
        for (int t = 0; t < 4; ++t)
#pragma unroll
            for (int r = 0; r < 8; ++r) {
                int m = m0 + a * 16 + kh * 8 + r;
                size_t o = ((size_t)b * CC + m) * HWN + n0 + t * 16 + mr;
                outp[o] = xq[o] + acc[a][t][r];
            }
}

// ---------------------------------------------------------------------------
extern "C" void kernel_launch(void* const* d_in, const int* in_sizes, int n_in,
                              void* d_out, int out_size, void* d_ws, size_t ws_size,
                              hipStream_t stream) {
    const float* x_q  = (const float*)d_in[0];
    const float* x_kv = (const float*)d_in[1];
    const float* wq   = (const float*)d_in[2];
    const float* bq   = (const float*)d_in[3];
    const float* wk   = (const float*)d_in[4];
    const float* bk   = (const float*)d_in[5];
    const float* wv   = (const float*)d_in[6];
    const float* bv   = (const float*)d_in[7];

    float* out      = (float*)d_out;
    float* out_x    = out;                               // (B,C,H,W)
    float* out_attn = out + (size_t)BB * CC * HWN;       // (B,C,C)

    char* ws = (char*)d_ws;
    const size_t S = (size_t)BB * CC * HWN * sizeof(__bf16);   // 33.5 MB
    __bf16* qb    = (__bf16*)(ws);                              // (B,C,HW)  bf16
    __bf16* kb    = (__bf16*)(ws + S);                          // (B,C,HW)  bf16
    __bf16* vT    = (__bf16*)(ws + 2 * S);                      // (B,HW,C)  bf16
    float*  attn  = (float*)(ws + 3 * S);                       // (B,C,C)   f32
    __bf16* attnm = (__bf16*)(ws + 3 * S + (size_t)BB * CC * CC * sizeof(float));

    const int total = BB * CC * HWN;
    dim3 cblk(256), cgrd((total + 255) / 256);
    dwconv3x3_bf16<<<cgrd, cblk, 0, stream>>>(x_q,  wq, bq, qb, 0);
    dwconv3x3_bf16<<<cgrd, cblk, 0, stream>>>(x_kv, wk, bk, kb, 0);
    dwconv3x3_bf16<<<cgrd, cblk, 0, stream>>>(x_kv, wv, bv, vT, 1);

    gemm_qk_wmma<<<dim3(CC / 64, CC / 32, BB), 32, 0, stream>>>(qb, kb, attn);
    softmax_rows<<<BB * CC, 32, 0, stream>>>(attn);
    topk_mask<<<dim3(CC, BB), 32, 0, stream>>>(attn, out_attn, attnm);
    gemm_av_wmma<<<dim3(HWN / 64, CC / 32, BB), 32, 0, stream>>>(attnm, vT, x_q, out_x);
}
